// mLSTM_cell_27779848470989
// MI455X (gfx1250) — matrix operand order repaired and checked by reading, not compiled, not run on hardware
//
#include <hip/hip_runtime.h>

#define DIM   512
#define HEADS 8
#define HD    64
#define SEQ   2048
#define BS    2
#define NBH   (BS*HEADS)      // 16
#define NTOK  (NBH*SEQ)       // 32768

typedef __attribute__((ext_vector_type(16))) _Float16 v16h;
typedef __attribute__((ext_vector_type(8)))  _Float16 v8h;
typedef __attribute__((ext_vector_type(8)))  float    v8f;

// ---------------------------------------------------------------------------
// Kernel 1: gate projections  itilde/ftilde[b,h,s] = dot(concat(q,k,v), W[h]) + b
// One wave per (b,h,s) token; lane-parallel over the 1536 features.
// ---------------------------------------------------------------------------
__global__ __launch_bounds__(256) void gates_kernel(
    const float* __restrict__ q, const float* __restrict__ k, const float* __restrict__ v,
    const float* __restrict__ Wi_w, const float* __restrict__ Wi_b,
    const float* __restrict__ Wf_w, const float* __restrict__ Wf_b,
    float* __restrict__ it_ws, float* __restrict__ ft_ws)
{
  int wave = (blockIdx.x * blockDim.x + threadIdx.x) >> 5;
  int lane = threadIdx.x & 31;
  int b   = wave / (HEADS * SEQ);
  int rem = wave % (HEADS * SEQ);
  int h   = rem / SEQ;
  int s   = rem % SEQ;

  const float* qrow = q + ((size_t)(b * SEQ + s)) * DIM;
  const float* krow = k + ((size_t)(b * SEQ + s)) * DIM;
  const float* vrow = v + ((size_t)(b * SEQ + s)) * DIM;
  const float* wi = Wi_w + (size_t)h * 3 * DIM;
  const float* wf = Wf_w + (size_t)h * 3 * DIM;

  float ai = 0.f, af = 0.f;
  for (int e = lane; e < DIM; e += 32) {
    float gq = qrow[e], gk = krow[e], gv = vrow[e];
    ai += gq * wi[e] + gk * wi[DIM + e] + gv * wi[2 * DIM + e];
    af += gq * wf[e] + gk * wf[DIM + e] + gv * wf[2 * DIM + e];
  }
#pragma unroll
  for (int off = 16; off; off >>= 1) {
    ai += __shfl_xor(ai, off);
    af += __shfl_xor(af, off);
  }
  if (lane == 0) {
    size_t o = (size_t)(b * HEADS + h) * SEQ + s;
    it_ws[o] = ai + Wi_b[h];
    ft_ws[o] = af + Wf_b[h];
  }
}

// ---------------------------------------------------------------------------
// Kernel 2: sequential scan per (b,h):
//   a_s = cumsum(logsigmoid(f)),  c_s = i_s - a_s,  M_s = prefixmax(c),
//   e_s = exp(-(a_s + M_s))  ( = exp(-max_d) )
// ---------------------------------------------------------------------------
__global__ void scan_kernel(const float* __restrict__ it_ws, const float* __restrict__ ft_ws,
                            float* __restrict__ c_ws, float* __restrict__ M_ws,
                            float* __restrict__ e_ws)
{
  int t = threadIdx.x;
  if (t >= NBH) return;
  const float* itb = it_ws + (size_t)t * SEQ;
  const float* ftb = ft_ws + (size_t)t * SEQ;
  float a = 0.f, Mv = -3.0e38f;
  for (int s = 0; s < SEQ; ++s) {
    float f  = ftb[s];
    float ls = fminf(f, 0.f) - log1pf(expf(-fabsf(f)));   // log_sigmoid
    a += ls;
    float cv = itb[s] - a;
    Mv = fmaxf(Mv, cv);
    size_t o = (size_t)t * SEQ + s;
    c_ws[o] = cv;
    M_ws[o] = Mv;
    e_ws[o] = expf(-(a + Mv));
  }
}

// ---------------------------------------------------------------------------
// Kernel 3: main mLSTM attention pass.
// Block = (b, h, 128-row block). 8 waves; wave w owns rows [i0, i0+16).
// j-chunks of 32 columns staged in LDS (f16) by the whole block.
//   S tile  : v_wmma_f32_16x16x32_f16 x2 (K = head_dim split 32+32)
//   Ctilde  : S * exp(c_j - M_i) / tau, causal-masked; rowsum accumulated
//   H tile  : v_wmma_f32_16x16x32_f16 x4 (K = 32 j's, N = 4 x 16 d's)
// Epilogue: denom = max(|rowsum|, exp(-max_d)) + 1e-6, then GroupNorm.
// ---------------------------------------------------------------------------
__global__ __launch_bounds__(256) void mlstm_main(
    const float* __restrict__ q, const float* __restrict__ k, const float* __restrict__ v,
    const float* __restrict__ c_ws, const float* __restrict__ M_ws, const float* __restrict__ e_ws,
    const float* __restrict__ gn_w, const float* __restrict__ gn_b,
    float* __restrict__ out)
{
  __shared__ __align__(32) _Float16 ldsK[32 * 64];      // [j][d]
  __shared__ __align__(32) _Float16 ldsV[64 * 32];      // [d][j]  (transposed)
  __shared__ __align__(32) _Float16 ldsC[8][16 * 32];   // per-wave C tile [row][j]

  const int RBLK = SEQ / 128;                 // 16 row-blocks
  int rb = blockIdx.x % RBLK;
  int h  = (blockIdx.x / RBLK) % HEADS;
  int b  = blockIdx.x / (RBLK * HEADS);
  int wave = threadIdx.x >> 5;
  int lane = threadIdx.x & 31;
  int m    = lane & 15;                       // column / row-within-tile index
  int ba8  = (lane < 16) ? 0 : 8;             // C/D row half select
  int bh16 = (lane < 16) ? 0 : 16;            // B half select
  int i0   = rb * 128 + wave * 16;
  int jend = rb * 128 + 128;
  size_t bh = (size_t)(b * HEADS + h);
  const float* cb = c_ws + bh * SEQ;

  // --- loop-invariant per-strip state -------------------------------------
  const float* qrow = q + ((size_t)(b * SEQ) + i0 + m) * DIM + h * HD;
  v16h aQ0, aQ1;                               // Q strip as WMMA A operands
#pragma unroll
  for (int p = 0; p < 8; ++p) {
    aQ0[p]     = (_Float16)qrow[bh16 + p];
    aQ0[p + 8] = (_Float16)qrow[bh16 + 8 + p];
    aQ1[p]     = (_Float16)qrow[32 + bh16 + p];
    aQ1[p + 8] = (_Float16)qrow[32 + bh16 + 8 + p];
  }
  float Mrow[8], erow[8], rowsum[8];
#pragma unroll
  for (int r = 0; r < 8; ++r) {
    Mrow[r]   = M_ws[bh * SEQ + i0 + ba8 + r];
    erow[r]   = e_ws[bh * SEQ + i0 + ba8 + r];
    rowsum[r] = 0.f;
  }
  v8f zero = {0.f, 0.f, 0.f, 0.f, 0.f, 0.f, 0.f, 0.f};
  v8f acc[4];
#pragma unroll
  for (int t = 0; t < 4; ++t) acc[t] = zero;

  // --- streaming causal j loop --------------------------------------------
  for (int jc = 0; jc < jend; jc += 32) {
    __syncthreads();  // previous chunk's LDS reads complete before overwrite
    {   // cooperative K/V chunk load, f32 -> f16, 8 elems per thread
      int eb  = threadIdx.x * 8;
      int row = eb >> 6;                       // 0..31 (j within chunk)
      int col = eb & 63;                       // 0..56 (d)
      size_t g = ((size_t)(b * SEQ) + jc + row) * DIM + h * HD + col;
      const float4* kp = (const float4*)(k + g);
      const float4* vp = (const float4*)(v + g);
      float4 k0 = kp[0], k1 = kp[1];
      float4 v0 = vp[0], v1 = vp[1];
      _Float16* kd = ldsK + row * 64 + col;
      kd[0] = (_Float16)k0.x; kd[1] = (_Float16)k0.y;
      kd[2] = (_Float16)k0.z; kd[3] = (_Float16)k0.w;
      kd[4] = (_Float16)k1.x; kd[5] = (_Float16)k1.y;
      kd[6] = (_Float16)k1.z; kd[7] = (_Float16)k1.w;
      ldsV[(col + 0) * 32 + row] = (_Float16)v0.x;
      ldsV[(col + 1) * 32 + row] = (_Float16)v0.y;
      ldsV[(col + 2) * 32 + row] = (_Float16)v0.z;
      ldsV[(col + 3) * 32 + row] = (_Float16)v0.w;
      ldsV[(col + 4) * 32 + row] = (_Float16)v1.x;
      ldsV[(col + 5) * 32 + row] = (_Float16)v1.y;
      ldsV[(col + 6) * 32 + row] = (_Float16)v1.z;
      ldsV[(col + 7) * 32 + row] = (_Float16)v1.w;
    }
    __syncthreads();

    if (jc > i0 + 15) continue;                // wave-uniform causal skip

    float cj0 = cb[jc + m];
    float cj1 = cb[jc + 16 + m];

#pragma unroll
    for (int jt = 0; jt < 2; ++jt) {
      // B operand: K^T tile, lane = column j, 16 contiguous K(d) halves
      v16h b0 = *(const v16h*)(ldsK + (jt * 16 + m) * 64 + 0  + bh16);
      v16h b1 = *(const v16h*)(ldsK + (jt * 16 + m) * 64 + 32 + bh16);
      v8f s = zero;
      s = __builtin_amdgcn_wmma_f32_16x16x32_f16(false, aQ0, false, b0, (short)0, s, false, false);
      s = __builtin_amdgcn_wmma_f32_16x16x32_f16(false, aQ1, false, b1, (short)0, s, false, false);
      float cj = jt ? cj1 : cj0;
      int jg = jc + jt * 16 + m;
#pragma unroll
      for (int r = 0; r < 8; ++r) {
        int row = i0 + ba8 + r;
        float val = (jg <= row) ? s[r] * 0.125f * __expf(cj - Mrow[r]) : 0.f;
        rowsum[r] += val;
        ldsC[wave][(ba8 + r) * 32 + jt * 16 + m] = (_Float16)val;
      }
    }
    // within-wave LDS store->load ordering (cross-lane transpose)
    asm volatile("s_wait_dscnt 0x0" ::: "memory");

    v16h aC;   // Ctilde tile as A operand (16 rows x 32 j)
    {
      v8h lo = *(const v8h*)(&ldsC[wave][m * 32 + bh16]);
      v8h hi = *(const v8h*)(&ldsC[wave][m * 32 + bh16 + 8]);
#pragma unroll
      for (int p = 0; p < 8; ++p) { aC[p] = lo[p]; aC[p + 8] = hi[p]; }
    }
#pragma unroll
    for (int t = 0; t < 4; ++t) {
      // B operand: V tile, lane = column d, 16 contiguous K(j) halves
      v16h bV = *(const v16h*)(ldsV + (t * 16 + m) * 32 + bh16);
      acc[t] = __builtin_amdgcn_wmma_f32_16x16x32_f16(false, aC, false, bV, (short)0, acc[t], false, false);
    }
  }

  // --- epilogue: denom, GroupNorm, store ----------------------------------
  const float* gwp = gn_w + h * HD;
  const float* gbp = gn_b + h * HD;
#pragma unroll
  for (int r = 0; r < 8; ++r) {
    float rs = rowsum[r];
    rs += __shfl_xor(rs, 1); rs += __shfl_xor(rs, 2);
    rs += __shfl_xor(rs, 4); rs += __shfl_xor(rs, 8);
    float inv = 1.0f / (fmaxf(fabsf(rs), erow[r]) + 1e-6f);
    float x0 = acc[0][r] * inv, x1 = acc[1][r] * inv;
    float x2 = acc[2][r] * inv, x3 = acc[3][r] * inv;
    float sm = x0 + x1 + x2 + x3;
    float sq = x0 * x0 + x1 * x1 + x2 * x2 + x3 * x3;
    sm += __shfl_xor(sm, 1); sm += __shfl_xor(sm, 2);
    sm += __shfl_xor(sm, 4); sm += __shfl_xor(sm, 8);
    sq += __shfl_xor(sq, 1); sq += __shfl_xor(sq, 2);
    sq += __shfl_xor(sq, 4); sq += __shfl_xor(sq, 8);
    float mu   = sm * (1.0f / 64.0f);
    float var  = sq * (1.0f / 64.0f) - mu * mu;
    float rstd = rsqrtf(var + 1e-5f);
    int row = i0 + ba8 + r;
    float* orow = out + ((size_t)(b * SEQ) + row) * DIM + h * HD;
    orow[0  + m] = (x0 - mu) * rstd * gwp[0  + m] + gbp[0  + m];
    orow[16 + m] = (x1 - mu) * rstd * gwp[16 + m] + gbp[16 + m];
    orow[32 + m] = (x2 - mu) * rstd * gwp[32 + m] + gbp[32 + m];
    orow[48 + m] = (x3 - mu) * rstd * gwp[48 + m] + gbp[48 + m];
  }
}

// ---------------------------------------------------------------------------
extern "C" void kernel_launch(void* const* d_in, const int* in_sizes, int n_in,
                              void* d_out, int out_size, void* d_ws, size_t ws_size,
                              hipStream_t stream) {
  (void)in_sizes; (void)n_in; (void)out_size; (void)ws_size;
  const float* q    = (const float*)d_in[0];
  const float* k    = (const float*)d_in[1];
  const float* v    = (const float*)d_in[2];
  const float* Wi_w = (const float*)d_in[3];
  const float* Wi_b = (const float*)d_in[4];
  const float* Wf_w = (const float*)d_in[5];
  const float* Wf_b = (const float*)d_in[6];
  const float* gn_w = (const float*)d_in[7];
  const float* gn_b = (const float*)d_in[8];
  float* out = (float*)d_out;

  float* ws    = (float*)d_ws;
  float* it_ws = ws;             // NTOK
  float* ft_ws = ws + 1 * (size_t)NTOK;
  float* c_ws  = ws + 2 * (size_t)NTOK;
  float* M_ws  = ws + 3 * (size_t)NTOK;
  float* e_ws  = ws + 4 * (size_t)NTOK;

  gates_kernel<<<NTOK / 8, 256, 0, stream>>>(q, k, v, Wi_w, Wi_b, Wf_w, Wf_b, it_ws, ft_ws);
  scan_kernel<<<1, 32, 0, stream>>>(it_ws, ft_ws, c_ws, M_ws, e_ws);
  mlstm_main<<<BS * HEADS * (SEQ / 128), 256, 0, stream>>>(q, k, v, c_ws, M_ws, e_ws, gn_w, gn_b, out);
}
